// MoEFFN_88338887344194
// MI455X (gfx1250) — matrix operand order repaired and checked by reading, not compiled, not run on hardware
//
#include <hip/hip_runtime.h>

typedef __attribute__((ext_vector_type(16))) __bf16 v16bf;
typedef __attribute__((ext_vector_type(8)))  float  v8f;

#define T_TOK   2048
#define DDIM    1024
#define HDIM    2816
#define NE      8
#define HC      32            // K per WMMA (bf16)
#define MTILE   32            // tokens per workgroup (2 sub-tiles of 16)
#define BLK     512           // 16 waves
#define LISTSTRIDE 2048       // max tokens per expert (top-2 of 2048)

// ---------- helpers ----------
__device__ __forceinline__ unsigned int pack_bf16x2(float lo, float hi) {
  unsigned int r;
  asm("v_cvt_pk_bf16_f32 %0, %1, %2" : "=v"(r) : "v"(lo), "v"(hi));
  return r;
}

// ---------- kernel 0: zero output + expert counters ----------
__global__ void moe_zero(float* __restrict__ out, int n, int* __restrict__ counts) {
  int i = blockIdx.x * blockDim.x + threadIdx.x;
  if (i < n) out[i] = 0.f;
  if (blockIdx.x == 0 && threadIdx.x < NE) counts[threadIdx.x] = 0;
}

// ---------- kernel 1: gating (one wave32 per token) ----------
__global__ void moe_gate(const float* __restrict__ x, const float* __restrict__ wg,
                         int* __restrict__ counts, int* __restrict__ list,
                         float* __restrict__ listw) {
  const int wave = threadIdx.x >> 5;
  const int lane = threadIdx.x & 31;
  const int t = blockIdx.x * 8 + wave;
  if (t >= T_TOK) return;
  const float4* xr = (const float4*)(x + (size_t)t * DDIM);
  const float4* wg4 = (const float4*)wg;

  float acc[NE];
#pragma unroll
  for (int e = 0; e < NE; ++e) acc[e] = 0.f;
  for (int d4 = lane; d4 < DDIM / 4; d4 += 32) {
    float4 xv = xr[d4];
#pragma unroll
    for (int e = 0; e < NE; ++e) {
      float4 wv = wg4[e * (DDIM / 4) + d4];
      acc[e] += xv.x * wv.x + xv.y * wv.y + xv.z * wv.z + xv.w * wv.w;
    }
  }
#pragma unroll
  for (int e = 0; e < NE; ++e) {
#pragma unroll
    for (int off = 16; off > 0; off >>= 1) acc[e] += __shfl_xor(acc[e], off, 32);
  }
  if (lane == 0) {
    float mx = acc[0];
#pragma unroll
    for (int e = 1; e < NE; ++e) mx = fmaxf(mx, acc[e]);
    float p[NE];
#pragma unroll
    for (int e = 0; e < NE; ++e) p[e] = __expf(acc[e] - mx);
    int a1 = 0;
#pragma unroll
    for (int e = 1; e < NE; ++e) if (p[e] > p[a1]) a1 = e;
    int a2 = (a1 == 0) ? 1 : 0;
#pragma unroll
    for (int e = 0; e < NE; ++e) if (e != a1 && p[e] > p[a2]) a2 = e;
    float denom = p[a1] + p[a2] + 1e-20f;
    float w1 = p[a1] / denom, w2 = p[a2] / denom;
    int p0 = atomicAdd(&counts[a1], 1);
    list[a1 * LISTSTRIDE + p0] = t;  listw[a1 * LISTSTRIDE + p0] = w1;
    int p1 = atomicAdd(&counts[a2], 1);
    list[a2 * LISTSTRIDE + p1] = t;  listw[a2 * LISTSTRIDE + p1] = w2;
  }
}

// Lane's B-tile (32x16 bf16: lanes 0-15 hold K=0..15, lanes 16-31 K=16..31)
// from 16 consecutive fp32 -> 4x global_load_b128 + 8x v_cvt_pk_bf16_f32.
__device__ __forceinline__ void load_b_tile(const float* __restrict__ bp,
                                            unsigned int* __restrict__ u) {
  float4 q0 = *(const float4*)(bp);
  float4 q1 = *(const float4*)(bp + 4);
  float4 q2 = *(const float4*)(bp + 8);
  float4 q3 = *(const float4*)(bp + 12);
  u[0] = pack_bf16x2(q0.x, q0.y); u[1] = pack_bf16x2(q0.z, q0.w);
  u[2] = pack_bf16x2(q1.x, q1.y); u[3] = pack_bf16x2(q1.z, q1.w);
  u[4] = pack_bf16x2(q2.x, q2.y); u[5] = pack_bf16x2(q2.z, q2.w);
  u[6] = pack_bf16x2(q3.x, q3.y); u[7] = pack_bf16x2(q3.z, q3.w);
}

// ---------- kernel 2: per-expert SwiGLU FFN via WMMA, 32-token tiles ----------
__global__ __launch_bounds__(BLK) void moe_ffn(
    const float* __restrict__ x,  const float* __restrict__ w1,
    const float* __restrict__ w2, const float* __restrict__ w3,
    const int* __restrict__ counts, const int* __restrict__ list,
    const float* __restrict__ listw, float* __restrict__ out) {
  __shared__ unsigned int sxu[MTILE * (DDIM / 2)];   // 64 KB: X tile, bf16 packed
  __shared__ float        spart[32 * 256];           // 32 KB: GEMM1 partials
  __shared__ unsigned int shu[MTILE * (HC / 2)];     // 2 KB: h tile, bf16 packed
  __shared__ float        srw[MTILE];
  __shared__ int          stok[MTILE];

  const int e  = blockIdx.x >> 6;          // 64 tiles of 32 tokens
  const int mt = blockIdx.x & 63;
  const int count = counts[e];
  if (mt * MTILE >= count) return;

  const int tid  = threadIdx.x;
  const int wv   = tid >> 5;               // 0..15
  const int lane = tid & 31;
  const int m    = lane & 15;
  const int hi16 = (lane & 16) ? 1 : 0;

  if (tid < MTILE) {
    int i = mt * MTILE + tid;
    if (i < count) { stok[tid] = list[e * LISTSTRIDE + i]; srw[tid] = listw[e * LISTSTRIDE + i]; }
    else           { stok[tid] = 0;                        srw[tid] = 0.f; }
  }
  __syncthreads();

  // X tile fp32 -> bf16 into LDS (float4 loads, uint2 stores)
  for (int i = tid; i < MTILE * (DDIM / 4); i += BLK) {
    int row = i >> 8, c4 = i & 255;                 // 256 float4 per row
    float4 q = *(const float4*)(x + (size_t)stok[row] * DDIM + c4 * 4);
    uint2 o; o.x = pack_bf16x2(q.x, q.y); o.y = pack_bf16x2(q.z, q.w);
    *(uint2*)&sxu[row * 512 + c4 * 2] = o;
  }
  __syncthreads();

  const int mat = wv & 1;          // 0 -> W1 (silu), 1 -> W3 (gate)
  const int nt  = (wv >> 1) & 1;   // h n-tile
  const int kq  = wv >> 2;         // K quarter of D (0..3)
  const float* wsel = mat ? w3 : w1;
  const int dcol0 = wv * 64;       // GEMM2 output column base (16 waves x 64)

  v8f vzero = {};
  v8f y[8];                        // [dtile 0..3][token sub-tile 0..1]
#pragma unroll
  for (int i = 0; i < 8; ++i) y[i] = vzero;

  for (int hc = 0; hc < HDIM; hc += HC) {
    // ---- GEMM1: one weight B-tile feeds BOTH token sub-tiles (2 WMMAs) ----
    v8f c0 = vzero, c1 = vzero;
    const size_t browbase = ((size_t)(e * HDIM + hc + nt * 16 + m)) * DDIM;
#pragma unroll
    for (int s = 0; s < 8; ++s) {
      const int kbase = kq * 256 + s * 32;
      union { v16bf v; unsigned int u[8]; } A0, A1, B;
      load_b_tile(wsel + browbase + kbase + hi16 * 16, B.u);
      {
        int idx = m * 512 + (kbase >> 1) + hi16 * 8;          // rows 0..15
        uint4 alo = *(const uint4*)&sxu[idx];
        uint4 ahi = *(const uint4*)&sxu[idx + 4];
        A0.u[0] = alo.x; A0.u[1] = alo.y; A0.u[2] = alo.z; A0.u[3] = alo.w;
        A0.u[4] = ahi.x; A0.u[5] = ahi.y; A0.u[6] = ahi.z; A0.u[7] = ahi.w;
        idx += 16 * 512;                                       // rows 16..31
        alo = *(const uint4*)&sxu[idx];
        ahi = *(const uint4*)&sxu[idx + 4];
        A1.u[0] = alo.x; A1.u[1] = alo.y; A1.u[2] = alo.z; A1.u[3] = alo.w;
        A1.u[4] = ahi.x; A1.u[5] = ahi.y; A1.u[6] = ahi.z; A1.u[7] = ahi.w;
      }
      c0 = __builtin_amdgcn_wmma_f32_16x16x32_bf16(false, A0.v, false, B.v,
                                                   (short)0, c0, false, false);
      c1 = __builtin_amdgcn_wmma_f32_16x16x32_bf16(false, A1.v, false, B.v,
                                                   (short)0, c1, false, false);
    }
    {
      int roff = hi16 * 8;
#pragma unroll
      for (int r = 0; r < 8; ++r) {
        spart[(wv * 2 + 0) * 256 + (r + roff) * 16 + m] = c0[r];
        spart[(wv * 2 + 1) * 256 + (r + roff) * 16 + m] = c1[r];
      }
    }
    __syncthreads();

    // ---- combine K-quarters, SiLU * gate; one packed pair per thread ----
    {
      int pi  = tid;                   // pair index 0..511
      int t   = pi >> 4;               // token row 0..31
      int hp  = pi & 15;               // h pair -> hh = 2*hp, 2*hp+1
      int mt2 = t >> 4;                // token sub-tile
      int bnt = hp >> 3;               // h n-tile
      int off = (t & 15) * 16 + ((2 * hp) & 15);
      // buffer id = (mat + 2*nt + 4*kq)*2 + mt2
      float p1a = 0.f, p3a = 0.f, p1b = 0.f, p3b = 0.f;
#pragma unroll
      for (int q = 0; q < 4; ++q) {
        int b1 = ((0 + 2 * bnt + 4 * q) * 2 + mt2) * 256 + off;
        int b3 = ((1 + 2 * bnt + 4 * q) * 2 + mt2) * 256 + off;
        p1a += spart[b1];     p3a += spart[b3];
        p1b += spart[b1 + 1]; p3b += spart[b3 + 1];
      }
      float hv0 = (p1a / (1.f + __expf(-p1a))) * p3a;
      float hv1 = (p1b / (1.f + __expf(-p1b))) * p3b;
      shu[t * 16 + hp] = pack_bf16x2(hv0, hv1);
    }
    __syncthreads();

    // ---- GEMM2: one W2 B-tile feeds BOTH token sub-tiles (2 WMMAs) ----
    union { v16bf v; unsigned int u[8]; } A2a, A2b;
    {
      int idx = m * 16 + hi16 * 8;                    // rows 0..15
      uint4 alo = *(const uint4*)&shu[idx];
      uint4 ahi = *(const uint4*)&shu[idx + 4];
      A2a.u[0] = alo.x; A2a.u[1] = alo.y; A2a.u[2] = alo.z; A2a.u[3] = alo.w;
      A2a.u[4] = ahi.x; A2a.u[5] = ahi.y; A2a.u[6] = ahi.z; A2a.u[7] = ahi.w;
      idx += 16 * 16;                                  // rows 16..31
      alo = *(const uint4*)&shu[idx];
      ahi = *(const uint4*)&shu[idx + 4];
      A2b.u[0] = alo.x; A2b.u[1] = alo.y; A2b.u[2] = alo.z; A2b.u[3] = alo.w;
      A2b.u[4] = ahi.x; A2b.u[5] = ahi.y; A2b.u[6] = ahi.z; A2b.u[7] = ahi.w;
    }
#pragma unroll
    for (int i = 0; i < 4; ++i) {
      union { v16bf v; unsigned int u[8]; } B2;
      const float* bp0 =
          w2 + ((size_t)(e * DDIM + dcol0 + i * 16 + m)) * HDIM + hc + hi16 * 16;
      load_b_tile(bp0, B2.u);
      y[2 * i + 0] = __builtin_amdgcn_wmma_f32_16x16x32_bf16(false, A2a.v, false, B2.v,
                                                             (short)0, y[2 * i + 0], false, false);
      y[2 * i + 1] = __builtin_amdgcn_wmma_f32_16x16x32_bf16(false, A2b.v, false, B2.v,
                                                             (short)0, y[2 * i + 1], false, false);
    }
  }

  // ---- weighted scatter-add (exactly 2 contributions per element -> deterministic) ----
  const int roff = hi16 * 8;
#pragma unroll
  for (int i = 0; i < 4; ++i) {
    int col = dcol0 + i * 16 + m;
#pragma unroll
    for (int mt2 = 0; mt2 < 2; ++mt2) {
#pragma unroll
      for (int r = 0; r < 8; ++r) {
        int row = mt2 * 16 + r + roff;
        float wgt = srw[row];
        if (wgt != 0.f)
          atomicAdd(out + (size_t)stok[row] * DDIM + col, wgt * y[2 * i + mt2][r]);
      }
    }
  }
}

extern "C" void kernel_launch(void* const* d_in, const int* in_sizes, int n_in,
                              void* d_out, int out_size, void* d_ws, size_t ws_size,
                              hipStream_t stream) {
  (void)in_sizes; (void)n_in; (void)ws_size;
  const float* x  = (const float*)d_in[0];
  const float* wg = (const float*)d_in[1];
  const float* w1 = (const float*)d_in[2];
  const float* w2 = (const float*)d_in[3];
  const float* w3 = (const float*)d_in[4];
  float* out = (float*)d_out;

  int*   counts = (int*)d_ws;
  int*   list   = (int*)((char*)d_ws + 256);
  float* listw  = (float*)((char*)d_ws + 256 + (size_t)NE * LISTSTRIDE * sizeof(int));

  moe_zero<<<(out_size + 255) / 256, 256, 0, stream>>>(out, out_size, counts);
  moe_gate<<<T_TOK / 8, 256, 0, stream>>>(x, wg, counts, list, listw);
  moe_ffn<<<NE * (T_TOK / MTILE), BLK, 0, stream>>>(x, w1, w2, w3, counts, list, listw, out);
}
